// EncoderBlock_16647293239647
// MI455X (gfx1250) — matrix operand order repaired and checked by reading, not compiled, not run on hardware
//
#include <hip/hip_runtime.h>

// ---------------------------------------------------------------------------
// Transformer encoder block for MI455X (gfx1250): wave32 + v_wmma_f32_16x16x32_f16
// GEMMs register-block 32x64 per wave (8 WMMA / 12 b128 loads per k-step).
// ---------------------------------------------------------------------------

typedef __attribute__((ext_vector_type(16))) _Float16 v16h;
typedef __attribute__((ext_vector_type(8)))  _Float16 v8h;
typedef __attribute__((ext_vector_type(8)))  float    v8f;

#define BATCH 2
#define SEQ   2048
#define DM    512
#define NH    8
#define DK    64
#define DFF   2048
#define NEGMAX (-1e30f)

// --- WMMA operand loaders -----------------------------------------------------
// A-operand (16x32 f16): lane l holds row m=l&15; element e -> K = e + 8*(e>=8) + 8*(l>=16)
// => two contiguous 8-half (16B) runs at offsets {8*hi, 16+8*hi}.
static __device__ __forceinline__ v16h load_a32(const _Float16* p, int hi) {
    v8h lo = *(const v8h*)(p + 8 * hi);
    v8h hs = *(const v8h*)(p + 16 + 8 * hi);
    v16h r;
#pragma unroll
    for (int i = 0; i < 8; ++i) { r[i] = lo[i]; r[i + 8] = hs[i]; }
    return r;
}
// B-operand (32x16 f16): lane l holds column n=l&15; element e -> K = e + 8*(e>=8) + 8*(l>=16)
// => two contiguous 8-half (16B) runs at offsets {8*hi, 16+8*hi} (requires K-contiguous storage).
static __device__ __forceinline__ v16h load_b32(const _Float16* p, int hi) {
    v8h lo = *(const v8h*)(p + 8 * hi);
    v8h hs = *(const v8h*)(p + 16 + 8 * hi);
    v16h r;
#pragma unroll
    for (int i = 0; i < 8; ++i) { r[i] = lo[i]; r[i + 8] = hs[i]; }
    return r;
}

static __device__ __forceinline__ v8f wmma_f16(v16h a, v16h b, v8f c) {
    return __builtin_amdgcn_wmma_f32_16x16x32_f16(false, a, false, b, (short)0, c,
                                                  false, false);
}

// --- fp32 weight [K x N] -> f16 transposed WT [N x K] -------------------------
__global__ __launch_bounds__(256) void pack_wt_kernel(const float* __restrict__ W,
                                                      _Float16* __restrict__ WT,
                                                      int K, int N) {
    int idx = blockIdx.x * 256 + threadIdx.x;
    if (idx < K * N) {
        int n = idx / K;
        int k = idx - n * K;
        WT[idx] = (_Float16)W[(size_t)k * N + n];
    }
}

// --- LayerNorm (torch-style: unbiased var, eps added to std), fp32 -> f16 -----
__global__ __launch_bounds__(256) void ln_kernel(const float* __restrict__ x,
                                                 _Float16* __restrict__ y,
                                                 const float* __restrict__ alpha,
                                                 const float* __restrict__ beta) {
    __shared__ float s1[256], s2[256];
    const int row = blockIdx.x, tid = threadIdx.x;
    const float* xr = x + (size_t)row * DM;
    float a = xr[tid], c = xr[tid + 256];
    s1[tid] = a + c;
    s2[tid] = a * a + c * c;
    __syncthreads();
    for (int off = 128; off > 0; off >>= 1) {
        if (tid < off) { s1[tid] += s1[tid + off]; s2[tid] += s2[tid + off]; }
        __syncthreads();
    }
    float mean  = s1[0] * (1.0f / DM);
    float var   = (s2[0] - (float)DM * mean * mean) * (1.0f / (DM - 1));
    float scale = alpha[0] / (sqrtf(var) + 1e-6f);
    float shift = beta[0];
    y[(size_t)row * DM + tid]       = (_Float16)((a - mean) * scale + shift);
    y[(size_t)row * DM + tid + 256] = (_Float16)((c - mean) * scale + shift);
}

// --- WMMA GEMM: C[M,N] = act(A[M,K] @ W + bias), 32x64 tile per wave ----------
// A row-major f16; W given as WT[N,K] f16.
// mode 0: fp32 out + residual   mode 1: f16 row-major (optional relu)
// mode 2: f16 Q/K layout [B,H,S,dk]   mode 3: f16 V-transposed [B,H,dk,S]
__global__ __launch_bounds__(256) void gemm_f16_kernel(
    const _Float16* __restrict__ A, const _Float16* __restrict__ WT,
    const float* __restrict__ bias, const float* __restrict__ residual,
    float* __restrict__ Cf, _Float16* __restrict__ Ch,
    int M, int N, int K, int mode, int relu) {
    const int lane = threadIdx.x & 31;
    const int wave = threadIdx.x >> 5;
    const int ntN  = N >> 6;                     // 64-col supertiles
    const int tile = blockIdx.x * 8 + wave;
    if (tile >= (M >> 5) * ntN) return;          // uniform per-wave
    const int tm = tile / ntN, tn = tile - tm * ntN;
    const int l15 = lane & 15, hi = lane >> 4;
    const int m0 = tm * 32, n0 = tn * 64;

    const _Float16* ar0 = A + (size_t)(m0 + l15) * K;
    const _Float16* ar1 = A + (size_t)(m0 + 16 + l15) * K;
    const _Float16* bc0 = WT + (size_t)(n0 + l15) * K;
    const _Float16* bc1 = WT + (size_t)(n0 + 16 + l15) * K;
    const _Float16* bc2 = WT + (size_t)(n0 + 32 + l15) * K;
    const _Float16* bc3 = WT + (size_t)(n0 + 48 + l15) * K;

    v8f acc[2][4];
    const v8f vzero = {};
#pragma unroll
    for (int i = 0; i < 2; ++i)
#pragma unroll
        for (int j = 0; j < 4; ++j) acc[i][j] = vzero;

    for (int k = 0; k < K; k += 32) {
        // speculative prefetch two k-blocks ahead (global_prefetch_b8)
        __builtin_prefetch(ar0 + k + 64, 0, 1);
        __builtin_prefetch(bc0 + k + 64, 0, 1);
        v16h a0 = load_a32(ar0 + k, hi);
        v16h a1 = load_a32(ar1 + k, hi);
        v16h b0 = load_b32(bc0 + k, hi);
        v16h b1 = load_b32(bc1 + k, hi);
        v16h b2 = load_b32(bc2 + k, hi);
        v16h b3 = load_b32(bc3 + k, hi);
        acc[0][0] = wmma_f16(a0, b0, acc[0][0]);
        acc[0][1] = wmma_f16(a0, b1, acc[0][1]);
        acc[0][2] = wmma_f16(a0, b2, acc[0][2]);
        acc[0][3] = wmma_f16(a0, b3, acc[0][3]);
        acc[1][0] = wmma_f16(a1, b0, acc[1][0]);
        acc[1][1] = wmma_f16(a1, b1, acc[1][1]);
        acc[1][2] = wmma_f16(a1, b2, acc[1][2]);
        acc[1][3] = wmma_f16(a1, b3, acc[1][3]);
    }

#pragma unroll
    for (int i = 0; i < 2; ++i) {
#pragma unroll
        for (int j = 0; j < 4; ++j) {
            const int n = n0 + j * 16 + l15;
            const float bn = bias[n];
#pragma unroll
            for (int r = 0; r < 8; ++r) {
                const int m = m0 + i * 16 + r + 8 * hi;
                float v = acc[i][j][r] + bn;
                if (relu) v = fmaxf(v, 0.0f);
                if (mode == 0) {
                    Cf[(size_t)m * N + n] = v + residual[(size_t)m * N + n];
                } else if (mode == 1) {
                    Ch[(size_t)m * N + n] = (_Float16)v;
                } else if (mode == 2) {  // [B,H,S,dk]
                    int b = m >> 11, s = m & (SEQ - 1);
                    int h = n >> 6,  d = n & (DK - 1);
                    Ch[(((size_t)(b * NH + h)) * SEQ + s) * DK + d] = (_Float16)v;
                } else {                  // [B,H,dk,S]
                    int b = m >> 11, s = m & (SEQ - 1);
                    int h = n >> 6,  d = n & (DK - 1);
                    Ch[(((size_t)(b * NH + h)) * DK + d) * SEQ + s] = (_Float16)v;
                }
            }
        }
    }
}

// --- Flash attention: one 16-query tile per wave, 32 keys per iteration -------
// Q,K: [B,H,S,dk] f16; Vt: [B,H,dk,S] f16; out: [B,S,DM] f16 (head-concat).
__global__ __launch_bounds__(256) void attn_kernel(
    const _Float16* __restrict__ Q, const _Float16* __restrict__ Kmat,
    const _Float16* __restrict__ Vt, const int* __restrict__ mask,
    _Float16* __restrict__ out) {
    __shared__ _Float16 plds[8 * 16 * 32];  // per-wave 16x32 P staging (8KB)
    const int lane = threadIdx.x & 31;
    const int wave = threadIdx.x >> 5;
    const int l15 = lane & 15, hi = lane >> 4;
    const int bh = blockIdx.y;
    const int b = bh >> 3, h = bh & 7;
    const int qt = blockIdx.x * 8 + wave;

    const _Float16* Qb = Q    + (size_t)bh * SEQ * DK;
    const _Float16* Kb = Kmat + (size_t)bh * SEQ * DK;
    const _Float16* Vb = Vt   + (size_t)bh * DK * SEQ;
    const int* mb = mask + b * SEQ;

    // Q A-operand, K-dim 64 = two 32-chunks; loaded once.
    const _Float16* qrow = Qb + (size_t)(qt * 16 + l15) * DK;
    v16h qa0 = load_a32(qrow, hi);
    v16h qa1 = load_a32(qrow + 32, hi);

    float mrow[8], lrow[8];
    v8f oacc[4];
    const v8f vzero = {};
#pragma unroll
    for (int r = 0; r < 8; ++r) { mrow[r] = NEGMAX; lrow[r] = 0.0f; }
#pragma unroll
    for (int dt = 0; dt < 4; ++dt) oacc[dt] = vzero;

    _Float16* pw = plds + wave * 512;

    for (int k0 = 0; k0 < SEQ; k0 += 32) {
        // ---- scores: two 16-key sub-tiles, QK^T over K-dim 64 ----
        float sc[2][8];
#pragma unroll
        for (int sub = 0; sub < 2; ++sub) {
            const int key = k0 + sub * 16 + l15;
            const _Float16* krow = Kb + (size_t)key * DK;
            v8f s = vzero;
            s = wmma_f16(qa0, load_b32(krow, hi), s);
            s = wmma_f16(qa1, load_b32(krow + 32, hi), s);
            const bool mk = (mb[key] != 0);
#pragma unroll
            for (int r = 0; r < 8; ++r)
                sc[sub][r] = mk ? s[r] * 0.125f : NEGMAX;  // 1/sqrt(64)
        }
        // ---- online softmax (row = r + 8*hi, cols spread across 16 lanes) ----
        float p0[8], p1[8];
#pragma unroll
        for (int r = 0; r < 8; ++r) {
            float mx = fmaxf(sc[0][r], sc[1][r]);
            mx = fmaxf(mx, __shfl_xor(mx, 1, 32));
            mx = fmaxf(mx, __shfl_xor(mx, 2, 32));
            mx = fmaxf(mx, __shfl_xor(mx, 4, 32));
            mx = fmaxf(mx, __shfl_xor(mx, 8, 32));
            float mnew = fmaxf(mrow[r], mx);
            float corr = __expf(mrow[r] - mnew);
            float ea = __expf(sc[0][r] - mnew);
            float eb = __expf(sc[1][r] - mnew);
            p0[r] = ea; p1[r] = eb;
            float ps = ea + eb;
            ps += __shfl_xor(ps, 1, 32);
            ps += __shfl_xor(ps, 2, 32);
            ps += __shfl_xor(ps, 4, 32);
            ps += __shfl_xor(ps, 8, 32);
            lrow[r] = lrow[r] * corr + ps;
            mrow[r] = mnew;
#pragma unroll
            for (int dt = 0; dt < 4; ++dt) oacc[dt][r] *= corr;
        }
        // ---- C-layout -> A-layout for P via wave-private LDS tile ----
        asm volatile("" ::: "memory");
#pragma unroll
        for (int r = 0; r < 8; ++r) {
            pw[(r + 8 * hi) * 32 + l15]      = (_Float16)p0[r];
            pw[(r + 8 * hi) * 32 + 16 + l15] = (_Float16)p1[r];
        }
        asm volatile("s_wait_dscnt 0" ::: "memory");  // wave-local DS ordering
        v16h pa = load_a32(pw + l15 * 32, hi);
        // ---- P[16x32] @ V[32x64]: 4 WMMAs across dk ----
#pragma unroll
        for (int dt = 0; dt < 4; ++dt) {
            const _Float16* vrow = Vb + (size_t)(dt * 16 + l15) * SEQ + k0;
            oacc[dt] = wmma_f16(pa, load_b32(vrow, hi), oacc[dt]);
        }
    }

    // ---- normalize and store head-concatenated f16 output ----
    float inv[8];
#pragma unroll
    for (int r = 0; r < 8; ++r) inv[r] = (lrow[r] > 0.0f) ? 1.0f / lrow[r] : 0.0f;
#pragma unroll
    for (int dt = 0; dt < 4; ++dt) {
#pragma unroll
        for (int r = 0; r < 8; ++r) {
            const int s   = qt * 16 + r + 8 * hi;
            const int col = h * DK + dt * 16 + l15;
            out[((size_t)(b * SEQ + s)) * DM + col] = (_Float16)(oacc[dt][r] * inv[r]);
        }
    }
}

// ---------------------------------------------------------------------------
extern "C" void kernel_launch(void* const* d_in, const int* in_sizes, int n_in,
                              void* d_out, int out_size, void* d_ws, size_t ws_size,
                              hipStream_t stream) {
    const float* x      = (const float*)d_in[0];
    const int*   mask   = (const int*)d_in[1];
    const float* Wq = (const float*)d_in[2];
    const float* bq = (const float*)d_in[3];
    const float* Wk = (const float*)d_in[4];
    const float* bk = (const float*)d_in[5];
    const float* Wv = (const float*)d_in[6];
    const float* bv = (const float*)d_in[7];
    const float* Wo = (const float*)d_in[8];
    const float* bo = (const float*)d_in[9];
    const float* W1 = (const float*)d_in[10];
    const float* b1 = (const float*)d_in[11];
    const float* W2 = (const float*)d_in[12];
    const float* b2 = (const float*)d_in[13];
    const float* alpha1 = (const float*)d_in[14];
    const float* beta1  = (const float*)d_in[15];
    const float* alpha2 = (const float*)d_in[16];
    const float* beta2  = (const float*)d_in[17];

    const int M = BATCH * SEQ;  // 4096 token rows
    char* ws = (char*)d_ws;
    // workspace layout (bytes), all 16B+ aligned
    _Float16* WqT = (_Float16*)(ws + 0);                 // 512 KB
    _Float16* WkT = (_Float16*)(ws + 524288);            // 512 KB
    _Float16* WvT = (_Float16*)(ws + 1048576);           // 512 KB
    _Float16* WoT = (_Float16*)(ws + 1572864);           // 512 KB
    _Float16* W1T = (_Float16*)(ws + 2097152);           // 2 MB
    _Float16* W2T = (_Float16*)(ws + 4194304);           // 2 MB
    _Float16* h1  = (_Float16*)(ws + 6291456);           // 4 MB (reused as h2)
    _Float16* Qf  = (_Float16*)(ws + 10485760);          // 4 MB
    _Float16* Kf  = (_Float16*)(ws + 14680064);          // 4 MB
    _Float16* Vt  = (_Float16*)(ws + 18874368);          // 4 MB
    _Float16* att = (_Float16*)(ws + 23068672);          // 4 MB
    float*    x1  = (float*)(ws + 27262976);             // 8 MB
    _Float16* a1  = Qf;   // 16 MB FFN act overlays dead Q/K/Vt/att region
    _Float16* h2  = h1;

    // ---- pack weights to f16 transposed ----
    pack_wt_kernel<<<(DM * DM + 255) / 256, 256, 0, stream>>>(Wq, WqT, DM, DM);
    pack_wt_kernel<<<(DM * DM + 255) / 256, 256, 0, stream>>>(Wk, WkT, DM, DM);
    pack_wt_kernel<<<(DM * DM + 255) / 256, 256, 0, stream>>>(Wv, WvT, DM, DM);
    pack_wt_kernel<<<(DM * DM + 255) / 256, 256, 0, stream>>>(Wo, WoT, DM, DM);
    pack_wt_kernel<<<(DM * DFF + 255) / 256, 256, 0, stream>>>(W1, W1T, DM, DFF);
    pack_wt_kernel<<<(DFF * DM + 255) / 256, 256, 0, stream>>>(W2, W2T, DFF, DM);

    // ---- LN1 ----
    ln_kernel<<<M, 256, 0, stream>>>(x, h1, alpha1, beta1);

    // ---- Q/K/V projections (fused bias + layout permute) ----
    const int wt_dm = (M / 32) * (DM / 64);    // 1024 wave-tiles -> 128 blocks
    gemm_f16_kernel<<<wt_dm / 8, 256, 0, stream>>>(h1, WqT, bq, nullptr, nullptr,
                                                   Qf, M, DM, DM, 2, 0);
    gemm_f16_kernel<<<wt_dm / 8, 256, 0, stream>>>(h1, WkT, bk, nullptr, nullptr,
                                                   Kf, M, DM, DM, 2, 0);
    gemm_f16_kernel<<<wt_dm / 8, 256, 0, stream>>>(h1, WvT, bv, nullptr, nullptr,
                                                   Vt, M, DM, DM, 3, 0);

    // ---- flash attention ----
    attn_kernel<<<dim3(SEQ / 128, BATCH * NH), 256, 0, stream>>>(Qf, Kf, Vt, mask, att);

    // ---- output projection + residual (fp32) ----
    gemm_f16_kernel<<<wt_dm / 8, 256, 0, stream>>>(att, WoT, bo, x, x1, nullptr,
                                                   M, DM, DM, 0, 0);

    // ---- LN2 + FFN ----
    ln_kernel<<<M, 256, 0, stream>>>(x1, h2, alpha2, beta2);
    const int wt_ff = (M / 32) * (DFF / 64);   // 4096 wave-tiles -> 512 blocks
    gemm_f16_kernel<<<wt_ff / 8, 256, 0, stream>>>(h2, W1T, b1, nullptr, nullptr,
                                                   a1, M, DFF, DM, 1, 1);
    gemm_f16_kernel<<<wt_dm / 8, 256, 0, stream>>>(a1, W2T, b2, x1, (float*)d_out,
                                                   nullptr, M, DM, DFF, 0, 0);
}
